// MultiFrameEstimatier_17755394801895
// MI455X (gfx1250) — compile-verified, matrix operand order checked by reading, not run on hardware
//
#include <hip/hip_runtime.h>
#include <hip/hip_bf16.h>

// CDNA5 / gfx1250: bf16 WMMA flash-attention pipeline (transposed-score form).
// wave32; WMMA 16x16x32 bf16 with f32 accumulation throughout.

typedef __bf16 bf16_t;
typedef bf16_t bf16x16 __attribute__((ext_vector_type(16)));
typedef bf16_t bf16x8  __attribute__((ext_vector_type(8)));
typedef float  f32x8   __attribute__((ext_vector_type(8)));

#define B_ 2
#define F_ 4
#define C_ 64
#define N_ 2048
#define H_ 4

// ---------------------------------------------------------------------------
// WMMA fragment helpers (wave32 layouts per CDNA5 ISA 7.12.2):
//  A (16x32 bf16): lane l -> row m=l&15, half kh=l>>4; VGPR j holds
//    K = (j>>2)*16 + kh*8 + (j&3)*2 (+1): two contiguous 16B chunks per lane.
//  B (32x16 bf16): mirror of A with n=l&15 as the column; loading from a
//    row-major [n][k] buffer uses the exact same pattern.
//  C/D (16x16 f32): lane l -> column n=l&15; VGPR r -> row m=(l>>4)*8+r.
// ---------------------------------------------------------------------------
__device__ __forceinline__ bf16x16 load_frag(const bf16_t* __restrict__ src, int ld) {
  const int lane = threadIdx.x & 31;
  const int m  = lane & 15;
  const int kh = (lane >> 4) & 1;
  const bf16_t* p = src + (size_t)m * ld + kh * 8;
  bf16x8 lo = *(const bf16x8*)(p);
  bf16x8 hi = *(const bf16x8*)(p + 16);
  bf16x16 r;
#pragma unroll
  for (int i = 0; i < 8; ++i) { r[i] = lo[i]; r[8 + i] = hi[i]; }
  return r;
}

__device__ __forceinline__ f32x8 wmma_bf16(bf16x16 a, bf16x16 b, f32x8 c) {
  return __builtin_amdgcn_wmma_f32_16x16x32_bf16(false, a, false, b, (short)0, c,
                                                 false, false);
}

// Pack two f32x8 C-tiles into a bf16 B-operand fragment.  With S computed
// transposed (lane = query, reg = key), this is layout-exact: B element i<8
// is key kh*8+i (tile0 reg i), element 8+i is key 16+kh*8+i (tile1 reg i).
__device__ __forceinline__ bf16x16 pack_pair(const f32x8 p0, const f32x8 p1) {
  bf16x16 r;
#pragma unroll
  for (int i = 0; i < 8; ++i) { r[i] = (bf16_t)p0[i]; r[8 + i] = (bf16_t)p1[i]; }
  return r;
}

// ---------------------------------------------------------------------------
// K0: bf16 weight conversion + zero the loss accumulator (runs every call so
// graph replay stays deterministic).
// ---------------------------------------------------------------------------
__global__ void k_init(const float* __restrict__ qw, const float* __restrict__ kvw,
                       const float* __restrict__ pw,
                       bf16_t* __restrict__ wq, bf16_t* __restrict__ wkv,
                       bf16_t* __restrict__ wproj, float* __restrict__ loss) {
  const int i = blockIdx.x * 256 + threadIdx.x;
  if (i == 0) *loss = 0.0f;
  if (i < 256 * 64) wq[i] = (bf16_t)qw[i];
  if (i < 512 * 64) wkv[i] = (bf16_t)kvw[i];
  if (i < 64 * 64) wproj[i] = (bf16_t)pw[i];
}

// ---------------------------------------------------------------------------
// K1: batchnorm + Q / K,V projections (WMMA).
// grid = B * F * (N/64); 4 waves, each owns 16 points.
// fsrc 0,1 -> Q for attention frame f=fsrc (scaled by C^-0.5).
// fsrc 3,2 -> K,V for attention frame f=3-fsrc (xr = xn reversed).
// Attention for output frames 2,3 is dead code in the reference; pruned.
// ---------------------------------------------------------------------------
__global__ __launch_bounds__(128) void k_bn_qkv(
    const float* __restrict__ xs, const float* __restrict__ gamma,
    const float* __restrict__ beta,
    const bf16_t* __restrict__ wq, const bf16_t* __restrict__ wkv,
    bf16_t* __restrict__ Qb, bf16_t* __restrict__ Kb, bf16_t* __restrict__ Vt) {
  __shared__ __align__(16) bf16_t xn[64][72];  // 144B rows: 16B-aligned frags
  int blk = blockIdx.x;
  const int nt = blk & 31; blk >>= 5;
  const int fsrc = blk & 3; const int b = blk >> 2;
  const int tid = threadIdx.x, wave = tid >> 5, lane = tid & 31;
  const float inv = rsqrtf(1.0f + 1e-5f);

  // Stage xn tile: xs is [b][f][c][n] (channel-major) -> LDS [point][channel].
  for (int idx = tid; idx < 64 * 64; idx += 128) {
    const int c = idx >> 6, nl = idx & 63;
    const float v =
        xs[(((size_t)b * F_ + fsrc) * C_ + c) * N_ + (size_t)nt * 64 + nl] * inv *
            gamma[c] + beta[c];
    xn[nl][c] = (bf16_t)v;
  }
  __syncthreads();

  const bf16_t* xrow = &xn[wave * 16][0];
  const bf16x16 aX0 = load_frag(xrow, 72);
  const bf16x16 aX1 = load_frag(xrow + 32, 72);
  const int n = lane & 15, hh = (lane >> 4) & 1;
  const int np0 = nt * 64 + wave * 16 + hh * 8;

  if (fsrc < 2) {                       // Q path, 256 output columns
    const int f = fsrc;
#pragma unroll 4
    for (int ot = 0; ot < 16; ++ot) {
      f32x8 acc = {};
      acc = wmma_bf16(aX0, load_frag(wq + ot * 16 * 64, 64), acc);
      acc = wmma_bf16(aX1, load_frag(wq + ot * 16 * 64 + 32, 64), acc);
      const int j = ot * 16 + n, h = j >> 6, c = j & 63;
      bf16_t* dst = Qb + ((((size_t)b * 2 + f) * H_ + h) * N_ + np0) * 64 + c;
#pragma unroll
      for (int r = 0; r < 8; ++r) dst[(size_t)r * 64] = (bf16_t)(acc[r] * 0.125f);
    }
  } else {                              // KV path, 512 output columns
    const int f = 3 - fsrc;
#pragma unroll 4
    for (int ot = 0; ot < 32; ++ot) {
      f32x8 acc = {};
      acc = wmma_bf16(aX0, load_frag(wkv + ot * 16 * 64, 64), acc);
      acc = wmma_bf16(aX1, load_frag(wkv + ot * 16 * 64 + 32, 64), acc);
      const int j = ot * 16 + n;
      if (ot < 16) {                    // K: row-major [n][c]
        const int h = j >> 6, c = j & 63;
        bf16_t* dst = Kb + ((((size_t)b * 2 + f) * H_ + h) * N_ + np0) * 64 + c;
#pragma unroll
        for (int r = 0; r < 8; ++r) dst[(size_t)r * 64] = (bf16_t)acc[r];
      } else {                          // V: stored transposed [c][n]
        const int j2 = j - 256, h = j2 >> 6, c = j2 & 63;
        bf16_t* dst = Vt + ((((size_t)b * 2 + f) * H_ + h) * 64 + c) * (size_t)N_ + np0;
#pragma unroll
        for (int r = 0; r < 8; ++r) dst[r] = (bf16_t)acc[r];
      }
    }
  }
}

// ---------------------------------------------------------------------------
// K2: fused flash attention in transposed form:  S^T = K.Q^T, O^T = V^T.P^T.
// grid = 16 heads * (N/64); one wave = 16 queries x 64 channels.
// Per 32-key block: 8 WMMAs; per-lane online softmax (lane = query, mirrored
// across the two 16-lane halves -> one shfl_xor(16) per reduction); the P
// operand is layout-exact from the S^T C-tiles (no LDS, no cross-lane moves).
// K-fragment loads are software-pipelined one block ahead.
// ---------------------------------------------------------------------------
__global__ __launch_bounds__(128) void k_attn(
    const bf16_t* __restrict__ Qb, const bf16_t* __restrict__ Kb,
    const bf16_t* __restrict__ Vt, float* __restrict__ Obuf) {
  int blk = blockIdx.x;
  const int qt = blk & 31; const int head = blk >> 5;  // head = ((b*2+f)*H + h)
  const int wave = threadIdx.x >> 5, lane = threadIdx.x & 31;
  const int n = lane & 15, hh = (lane >> 4) & 1;
  const int qrow = qt * 64 + wave * 16;

  const bf16_t* Qp = Qb + ((size_t)head * N_ + qrow) * 64;
  const bf16_t* Kp = Kb + (size_t)head * N_ * 64;
  const bf16_t* Vp = Vt + (size_t)head * 64 * N_;

  // Q^T as B operand: lane = query n, reads its own query row (contiguous).
  const bf16x16 bQ0 = load_frag(Qp, 64);
  const bf16x16 bQ1 = load_frag(Qp + 32, 64);

  f32x8 o0 = {}, o1 = {}, o2 = {}, o3 = {};   // O^T tiles: rows=channels
  float mcur = -1e30f, lcur = 0.0f;           // per-lane (= per-query) stats

  // Pipelined K A-fragments (rows = keys, K-dim = channels).
  bf16x16 a00 = load_frag(Kp, 64);
  bf16x16 a01 = load_frag(Kp + 32, 64);
  bf16x16 a10 = load_frag(Kp + 16 * 64, 64);
  bf16x16 a11 = load_frag(Kp + 16 * 64 + 32, 64);

  for (int kb = 0; kb < N_; kb += 32) {
    __builtin_prefetch(Kp + (size_t)(kb + 64) * 64, 0, 3);
    f32x8 s0 = {}, s1 = {};                   // S^T tiles: rows=keys, cols=queries
    s0 = wmma_bf16(a00, bQ0, s0);
    s0 = wmma_bf16(a01, bQ1, s0);
    s1 = wmma_bf16(a10, bQ0, s1);
    s1 = wmma_bf16(a11, bQ1, s1);

    // V^T A-fragments for this block (independent; overlap with softmax).
    const bf16x16 aV0 = load_frag(Vp + kb, N_);
    const bf16x16 aV1 = load_frag(Vp + (size_t)16 * N_ + kb, N_);
    const bf16x16 aV2 = load_frag(Vp + (size_t)32 * N_ + kb, N_);
    const bf16x16 aV3 = load_frag(Vp + (size_t)48 * N_ + kb, N_);

    // Next block's K fragments (branchless; tail read lands in adjacent ws).
    const bf16_t* Kn = Kp + (size_t)(kb + 32) * 64;
    a00 = load_frag(Kn, 64);
    a01 = load_frag(Kn + 32, 64);
    a10 = load_frag(Kn + 16 * 64, 64);
    a11 = load_frag(Kn + 16 * 64 + 32, 64);

    // Online softmax over this block's 32 keys (per lane: 16 local + mirror).
    float v = -1e30f;
#pragma unroll
    for (int r = 0; r < 8; ++r) v = fmaxf(v, fmaxf(s0[r], s1[r]));
    v = fmaxf(v, __shfl_xor(v, 16));
    const float mnew  = fmaxf(mcur, v);
    const float alpha = __expf(mcur - mnew);
    mcur = mnew;
    f32x8 p0 = {}, p1 = {};
    float ps = 0.0f;
#pragma unroll
    for (int r = 0; r < 8; ++r) {
      p0[r] = __expf(s0[r] - mnew);
      p1[r] = __expf(s1[r] - mnew);
      ps += p0[r] + p1[r];
    }
    ps += __shfl_xor(ps, 16);
    lcur = lcur * alpha + ps;
#pragma unroll
    for (int r = 0; r < 8; ++r) {
      o0[r] *= alpha; o1[r] *= alpha; o2[r] *= alpha; o3[r] *= alpha;
    }

    const bf16x16 bP = pack_pair(p0, p1);     // layout-exact, no data movement
    o0 = wmma_bf16(aV0, bP, o0);
    o1 = wmma_bf16(aV1, bP, o1);
    o2 = wmma_bf16(aV2, bP, o2);
    o3 = wmma_bf16(aV3, bP, o3);
  }

  // O^T store: lane n = query, channels t*16 + hh*8 + r -> contiguous 32B.
  const float invl = 1.0f / lcur;
  float* Od = Obuf + ((size_t)head * N_ + qrow + n) * 64 + hh * 8;
  f32x8 w0, w1, w2, w3;
#pragma unroll
  for (int r = 0; r < 8; ++r) {
    w0[r] = o0[r] * invl; w1[r] = o1[r] * invl;
    w2[r] = o2[r] * invl; w3[r] = o3[r] * invl;
  }
  *(f32x8*)(Od +  0) = w0;
  *(f32x8*)(Od + 16) = w1;
  *(f32x8*)(Od + 32) = w2;
  *(f32x8*)(Od + 48) = w3;
}

// ---------------------------------------------------------------------------
// K3: xa = o[f=0]+o[f=1]; y = xa @ proj_w^T + proj_b (WMMA);
// frames = y @ map_w^T + map_b.  h>=1 -> outputs; h==0 -> shifted cloud Aw.
// grid = B * H * (N/64), 4 waves of 16 rows.
// ---------------------------------------------------------------------------
__global__ __launch_bounds__(128) void k_proj(
    const float* __restrict__ Obuf, const bf16_t* __restrict__ wproj,
    const float* __restrict__ proj_b, const float* __restrict__ map_w,
    const float* __restrict__ map_b, const float* __restrict__ pc1,
    float* __restrict__ out_xa, float* __restrict__ out_fr,
    float* __restrict__ Aw) {
  __shared__ __align__(16) bf16_t xt[4][16][72];
  __shared__ float yt[4][16][65];
  int blk = blockIdx.x;
  const int qt = blk & 31; blk >>= 5;
  const int h = blk & 3; const int b = blk >> 2;
  const int wave = threadIdx.x >> 5, lane = threadIdx.x & 31;
  const int n = lane & 15, hh = (lane >> 4) & 1;
  const int np_base = qt * 64 + wave * 16;

  {  // xa tile -> bf16 LDS, A-operand layout (row-major [row][c])
    const int row = lane >> 1, ch = (lane & 1) * 32;
    const size_t base0 =
        ((((size_t)b * 2 + 0) * H_ + h) * N_ + np_base + row) * 64 + ch;
    const size_t base1 =
        ((((size_t)b * 2 + 1) * H_ + h) * N_ + np_base + row) * 64 + ch;
#pragma unroll
    for (int i = 0; i < 32; ++i)
      xt[wave][row][ch + i] = (bf16_t)(Obuf[base0 + i] + Obuf[base1 + i]);
  }
  __syncthreads();

  const bf16x16 aX0 = load_frag(&xt[wave][0][0], 72);
  const bf16x16 aX1 = load_frag(&xt[wave][0][0] + 32, 72);
#pragma unroll
  for (int t = 0; t < 4; ++t) {
    f32x8 acc = {};
    acc = wmma_bf16(aX0, load_frag(wproj + t * 16 * 64, 64), acc);
    acc = wmma_bf16(aX1, load_frag(wproj + t * 16 * 64 + 32, 64), acc);
    const float pb = proj_b[t * 16 + n];
#pragma unroll
    for (int r = 0; r < 8; ++r) {
      const int m = hh * 8 + r;
      const float val = acc[r] + pb;
      yt[wave][m][t * 16 + n] = val;
      if (h > 0)
        out_xa[(((size_t)b * 3 + (h - 1)) * N_ + np_base + m) * 64 + t * 16 + n] = val;
    }
  }
  __syncthreads();

  if (lane < 16) {                       // 64 -> 3 map per row
    const int m = lane, np = np_base + m;
#pragma unroll
    for (int j = 0; j < 3; ++j) {
      float f = map_b[j];
      for (int c = 0; c < 64; ++c) f += yt[wave][m][c] * map_w[j * 64 + c];
      if (h == 0)
        Aw[((size_t)b * N_ + np) * 3 + j] = pc1[((size_t)b * 3 + j) * N_ + np] + f;
      else
        out_fr[(((size_t)b * 3 + (h - 1)) * N_ + np) * 3 + j] = f;
    }
  }
}

// ---------------------------------------------------------------------------
// K4: chamfer loss.  d[n][m] = |a_n - b_m|^2 with a = pc1^T + frames0,
// b = pc1^T.  dir 0: per-n min over m; dir 1: per-m min over n.
// Other-side points cached in 24 KB LDS; means accumulated via f32 atomics.
// ---------------------------------------------------------------------------
__global__ __launch_bounds__(256) void k_chamfer(
    const float* __restrict__ Aw, const float* __restrict__ pc1,
    float* __restrict__ loss) {
  __shared__ float pts[N_ * 3];
  int blk = blockIdx.x;
  const int tile = blk & 7; blk >>= 3;
  const int b = blk & 1; const int dir = blk >> 1;
  const int tid = threadIdx.x;
  for (int idx = tid; idx < N_ * 3; idx += 256) {
    if (dir == 0) {
      const int m = idx / 3, j = idx - m * 3;
      pts[idx] = pc1[((size_t)b * 3 + j) * N_ + m];
    } else {
      pts[idx] = Aw[(size_t)b * N_ * 3 + idx];
    }
  }
  __syncthreads();
  const int nn = tile * 256 + tid;
  float ax, ay, az;
  if (dir == 0) {
    const float* a = Aw + ((size_t)b * N_ + nn) * 3;
    ax = a[0]; ay = a[1]; az = a[2];
  } else {
    ax = pc1[((size_t)b * 3 + 0) * N_ + nn];
    ay = pc1[((size_t)b * 3 + 1) * N_ + nn];
    az = pc1[((size_t)b * 3 + 2) * N_ + nn];
  }
  float mn = 1e30f;
  for (int m = 0; m < N_; ++m) {
    const float dx = ax - pts[m * 3 + 0];
    const float dy = ay - pts[m * 3 + 1];
    const float dz = az - pts[m * 3 + 2];
    mn = fminf(mn, dx * dx + dy * dy + dz * dz);
  }
  atomicAdd(loss, mn * (1.0f / (B_ * N_)));
}

// ---------------------------------------------------------------------------
// Workspace layout (bytes), all 16B-aligned.
// ---------------------------------------------------------------------------
static const size_t OFF_WQ    = 0;                        // 256*64 bf16
static const size_t OFF_WKV   = 32768;                    // 512*64 bf16
static const size_t OFF_WPROJ = 98304;                    // 64*64  bf16
static const size_t OFF_QB    = 131072;                   // [16][N][64] bf16
static const size_t OFF_KB    = OFF_QB + (size_t)16 * N_ * 64 * 2;
static const size_t OFF_VT    = OFF_KB + (size_t)16 * N_ * 64 * 2;
static const size_t OFF_OBUF  = OFF_VT + (size_t)16 * N_ * 64 * 2;
static const size_t OFF_AW    = OFF_OBUF + (size_t)16 * N_ * 64 * 4;

extern "C" void kernel_launch(void* const* d_in, const int* in_sizes, int n_in,
                              void* d_out, int out_size, void* d_ws, size_t ws_size,
                              hipStream_t stream) {
  (void)in_sizes; (void)n_in; (void)out_size; (void)ws_size;
  const float* xs     = (const float*)d_in[0];
  const float* pc1    = (const float*)d_in[1];
  // d_in[2] = pc2 (unused by the forward pass)
  const float* gamma  = (const float*)d_in[3];
  const float* beta   = (const float*)d_in[4];
  const float* q_w    = (const float*)d_in[5];
  const float* kv_w   = (const float*)d_in[6];
  const float* proj_w = (const float*)d_in[7];
  const float* proj_b = (const float*)d_in[8];
  const float* map_w  = (const float*)d_in[9];
  const float* map_b  = (const float*)d_in[10];

  char* ws = (char*)d_ws;
  bf16_t* wq    = (bf16_t*)(ws + OFF_WQ);
  bf16_t* wkv   = (bf16_t*)(ws + OFF_WKV);
  bf16_t* wproj = (bf16_t*)(ws + OFF_WPROJ);
  bf16_t* Qb    = (bf16_t*)(ws + OFF_QB);
  bf16_t* Kb    = (bf16_t*)(ws + OFF_KB);
  bf16_t* Vt    = (bf16_t*)(ws + OFF_VT);
  float*  Obuf  = (float*)(ws + OFF_OBUF);
  float*  Aw    = (float*)(ws + OFF_AW);

  float* out    = (float*)d_out;
  float* out_xa = out;                                   // (B,3,N,64)
  float* out_fr = out + (size_t)B_ * 3 * N_ * 64;        // (B,3,N,3)
  float* out_ls = out_fr + (size_t)B_ * 3 * N_ * 3;      // scalar

  k_init<<<128, 256, 0, stream>>>(q_w, kv_w, proj_w, wq, wkv, wproj, out_ls);
  k_bn_qkv<<<B_ * F_ * (N_ / 64), 128, 0, stream>>>(xs, gamma, beta, wq, wkv,
                                                    Qb, Kb, Vt);
  k_attn<<<16 * (N_ / 64), 128, 0, stream>>>(Qb, Kb, Vt, Obuf);
  k_proj<<<B_ * H_ * (N_ / 64), 128, 0, stream>>>(Obuf, wproj, proj_b, map_w,
                                                  map_b, pc1, out_xa, out_fr, Aw);
  k_chamfer<<<2 * B_ * (N_ / 256), 256, 0, stream>>>(Aw, pc1, out_ls);
}
